// ADAARNN_84301618086389
// MI455X (gfx1250) — matrix operand order repaired and checked by reading, not compiled, not run on hardware
//
#include <hip/hip_runtime.h>
#include <hip/hip_bf16.h>

// CDNA5 / gfx1250, wave32. Single workgroup (256 threads = 8 waves) runs the
// entire sequential scan. W_hh lives in VGPRs as bf16 WMMA A-operands, h lives
// in LDS as packed bf16 (B-operand, broadcast along N), z round-trips through
// LDS f32. Matvec per step = 4 chained v_wmma_f32_16x16x32_bf16 per wave.

typedef __attribute__((ext_vector_type(16))) __bf16 v16bf;
typedef __attribute__((ext_vector_type(8)))  float  v8f;

#define H 128
#define LOG2E 1.4426950408889634f
#define LN2   0.6931471805599453f
#define EPSC  1e-7f

__device__ __forceinline__ float fast_tanh(float v) {
    // tanh(v) = sign(v) * (1-e)/(1+e),  e = exp(-2|v|)
    float s = __builtin_fabsf(v);
    float e = __builtin_amdgcn_exp2f(-2.0f * LOG2E * s);
    float m = (1.0f - e) * __builtin_amdgcn_rcpf(1.0f + e);
    return __builtin_copysignf(m, v);
}

__global__ void __launch_bounds__(256, 1)
adaa_rnn_scan(const float* __restrict__ x,
              const float* __restrict__ h_in,
              const float* __restrict__ w_ih,
              const float* __restrict__ w_hh,
              const float* __restrict__ b_ih,
              const float* __restrict__ b_hh,
              float* __restrict__ out,   // [T*H states] ++ [H h_last]
              int T) {
    __shared__ __align__(32) __bf16 h_bf[H];   // current state, packed bf16 (B operand source)
    __shared__ __align__(32) float  zsh[H];    // z = W_hh @ h (f32)

    const int tid  = threadIdx.x;
    const int lane = tid & 31;
    const int wv   = tid >> 5;          // wave id 0..7 -> M-tile rows [16*wv, 16*wv+16)
    const int m    = lane & 15;
    const int hi   = (lane >> 4) & 1;   // lane half: selects K sub-range in A/B layouts
    const int row  = wv * 16 + m;       // this lane's A-matrix row

    // ---- Preload W_hh row into bf16 A-operand tiles (16x32 each, 4 K-tiles). ----
    // A layout used here (not ISA table order): lanes 0-15 / 16-31 both cover M=0..15;
// VGPR v: K = ktile*32 + hi*16 + 2v(+1) (contiguous, matches B).
    v16bf A[4];
#pragma unroll
    for (int kt = 0; kt < 4; ++kt) {
#pragma unroll
        for (int v = 0; v < 8; ++v) {
            int kb = kt * 32 + hi * 16 + 2 * v;
            float w0 = w_hh[row * H + kb];
            float w1 = w_hh[row * H + kb + 1];
            A[kt][2 * v]     = (__bf16)w0;
            A[kt][2 * v + 1] = (__bf16)w1;
        }
    }

    // ---- Per-lane nonlinearity constants (lanes 0..127 own state element tid). ----
    float wi = 0.0f, bsum = 0.0f, hcur = 0.0f;
    if (tid < H) {
        wi   = w_ih[tid];
        bsum = b_ih[tid] + b_hh[tid];
        hcur = h_in[tid];
        h_bf[tid] = (__bf16)hcur;
    }
    __syncthreads();

    // B-operand LDS byte offsets: B(32x16) bf16, lanes 0-15 hold K=0..15 (pairs per
    // VGPR), lanes 16-31 hold K=16..31 -> contiguous 16 bf16 per lane-half.
    const int boff = hi * 16;           // element offset into h_bf per K-tile
    const int zoff = wv * 16 + hi * 8;  // D layout: VGPR r <-> M = r + 8*hi

    float xi1 = x[T - 1];               // jnp.roll: step 0 pairs x[0] with x[T-1]

    for (int t = 0; t < T; ++t) {
        float xi = x[t];                               // uniform -> s_load
        if (t + 256 < T) __builtin_prefetch(x + t + 256, 0, 0);

        // ---- Stage 1 (all 256 lanes, EXEC full for WMMA): z-tile = W-tile @ h ----
        v8f acc = {};
#pragma unroll
        for (int kt = 0; kt < 4; ++kt) {
            v16bf B = *(const v16bf*)&h_bf[kt * 32 + boff];
            acc = __builtin_amdgcn_wmma_f32_16x16x32_bf16(
                false, A[kt], false, B, (short)0, acc, false, false);
        }
        // Every lane in a half holds identical 8 rows (N-broadcast) -> plain store.
        *(v8f*)&zsh[zoff] = acc;
        __syncthreads();

        // ---- Stage 2 (lanes 0..127): nonlinearity, state update, trace store ----
        if (tid < H) {
            float z = zsh[tid] + bsum;
            float a = __builtin_fmaf(wi, xi,  z);
            float c = __builtin_fmaf(wi, xi1, z);
            float diff = xi - xi1;                     // uniform
            float hn;
            if (__builtin_fabsf(diff) > EPSC) {        // uniform branch (hot path)
                // (logcosh(a)-logcosh(c))/diff ; the -log2 terms cancel
                float aa = __builtin_fabsf(a);
                float ac = __builtin_fabsf(c);
                float ea = __builtin_amdgcn_exp2f(-2.0f * LOG2E * aa);
                float ec = __builtin_amdgcn_exp2f(-2.0f * LOG2E * ac);
                float la = __builtin_amdgcn_logf(1.0f + ea);   // log2
                float lc = __builtin_amdgcn_logf(1.0f + ec);
                hn = ((aa - ac) + LN2 * (la - lc)) * __builtin_amdgcn_rcpf(diff);
            } else {
                hn = 0.5f * (fast_tanh(a) + fast_tanh(c));
            }
            hcur = hn;
            out[(size_t)t * H + tid] = hn;             // states trace
            h_bf[tid] = (__bf16)hn;                    // next step's B operand
        }
        __syncthreads();
        xi1 = xi;
    }

    if (tid < H) out[(size_t)T * H + tid] = hcur;      // h_last
}

extern "C" void kernel_launch(void* const* d_in, const int* in_sizes, int n_in,
                              void* d_out, int out_size, void* d_ws, size_t ws_size,
                              hipStream_t stream) {
    const float* x    = (const float*)d_in[0];
    const float* h    = (const float*)d_in[1];
    const float* w_ih = (const float*)d_in[2];
    const float* w_hh = (const float*)d_in[3];
    const float* b_ih = (const float*)d_in[4];
    const float* b_hh = (const float*)d_in[5];
    float* out = (float*)d_out;
    int T = in_sizes[0];   // x is (1,T,1) -> flat T

    adaa_rnn_scan<<<dim3(1), dim3(256), 0, stream>>>(x, h, w_ih, w_hh, b_ih, b_hh, out, T);
}
